// MHSA_19104014532754
// MI455X (gfx1250) — compile-verified, matrix operand order checked by reading, not run on hardware
//
#include <hip/hip_runtime.h>
#include <hip/hip_bf16.h>

typedef __bf16 bf16_t;
typedef __attribute__((ext_vector_type(16))) __bf16 v16bf;
typedef __attribute__((ext_vector_type(8)))  __bf16 v8bf;
typedef __attribute__((ext_vector_type(4)))  __bf16 v4bf;
typedef __attribute__((ext_vector_type(2)))  __bf16 v2bf;
typedef __attribute__((ext_vector_type(8)))  float  v8f;
typedef __attribute__((ext_vector_type(4)))  float  v4f;
typedef unsigned int u32;
typedef __attribute__((ext_vector_type(4))) u32 u32x4;
typedef __attribute__((ext_vector_type(8))) int i32x8;
typedef __attribute__((ext_vector_type(4))) int i32x4;

#define HEADS 8
#define DH    64          // head dim
#define NPIX  1024        // H*W
#define CCH   512         // channels
#define BATCH 16

// 16x32 bf16 WMMA operand fragment (A layout; B uses a transposed LDS tile so
// the same contiguous-K gather applies).  lane<16: K {0..7,16..23},
// lane>=16: K {8..15,24..31}.  Caller passes base = row_ptr + kb (kb = hi*8).
__device__ inline v16bf load_frag(const bf16_t* base) {
    v8bf lo = *(const v8bf*)(base);
    v8bf hi = *(const v8bf*)(base + 16);
    v16bf f;
#pragma unroll
    for (int i = 0; i < 8; ++i) { f[i] = lo[i]; f[8 + i] = hi[i]; }
    return f;
}

__device__ inline v8f wmma_bf16(v16bf a, v16bf b, v8f c) {
    return __builtin_amdgcn_wmma_f32_16x16x32_bf16(
        false, a, false, b, (short)0, c, false, false);
}

// Tensor Data Mover: 2D tile (global -> LDS), bf16 elements, LDS padding
// reproduces the bank-conflict-free strides.  D# per CDNA5 ISA.
//   pad_code_i: pad interval = 2^(code+1) dwords;  pad_code_a: pad = code+1 dw
__device__ inline void tdm_load_2d(u32 lds_addr, const bf16_t* gptr,
                                   u32 tdim0, u32 tdim1, u32 tile0, u32 tile1,
                                   u32 stride0, u32 pad_code_i, u32 pad_code_a)
{
    unsigned long long ga = (unsigned long long)(size_t)gptr;
    u32x4 g0;
    g0[0] = 1u;                                           // count=1, user mode
    g0[1] = lds_addr;                                     // LDS byte address
    g0[2] = (u32)ga;                                      // global_addr[31:0]
    g0[3] = (u32)((ga >> 32) & 0x1FFFFFFu) | (2u << 30);  // [56:32] | type=2
    i32x8 g1;
    g1[0] = (int)((1u << 16)                              // data_size = 2B
                | (1u << 20)                              // pad_enable
                | (pad_code_i << 22) | (pad_code_a << 25));
    g1[1] = (int)(tdim0 << 16);                           // tensor_dim0 lo16
    g1[2] = (int)((tdim0 >> 16) | (tdim1 << 16));         // dim0 hi | dim1 lo
    g1[3] = (int)((tdim1 >> 16) | (tile0 << 16));         // dim1 hi | tile0
    g1[4] = (int)tile1;                                   // tile1 | tile2(0)
    g1[5] = (int)stride0;                                 // dim0_stride lo32
    g1[6] = 0;                                            // stride hi | s1 lo
    g1[7] = 0;
    i32x4 z4 = {0, 0, 0, 0};
    i32x8 z8 = {0, 0, 0, 0, 0, 0, 0, 0};
    __builtin_amdgcn_tensor_load_to_lds(g0, g1, z4, z4, z8, 0);
}

// ---------------------------------------------------------------------------
// Fused QKV projection: one pass over x computes all three projections.
//   Q: *scale, layout [b][h][i][dd]      K: +(rel_h+rel_w), same layout
//   V: stored transposed [b][h][dd][i]
// Block: 256 threads (8 waves). Tile: 64(o) x 128(i), K-step 32, 12 WMMA/step.
// ---------------------------------------------------------------------------
#define SW_STRIDE 40      // 32 + 8 pad halfs: 80B, 16B-aligned, conflict-free
#define SO_STRIDE 72      // 64 + 8
#define SOV_STRIDE 136    // 128 + 8
__global__ __launch_bounds__(256) void qkv_proj_kernel(
    const float* __restrict__ x,
    const float* __restrict__ Wq, const float* __restrict__ bq,
    const float* __restrict__ Wk, const float* __restrict__ bk,
    const float* __restrict__ Wv, const float* __restrict__ bv,
    const float* __restrict__ rel_h, const float* __restrict__ rel_w,
    bf16_t* __restrict__ q_ws, bf16_t* __restrict__ k_ws,
    bf16_t* __restrict__ v_ws)
{
    __shared__ bf16_t sW[3][64 * SW_STRIDE];   // [o][c] per matrix
    __shared__ bf16_t sX[128 * SW_STRIDE];     // [i][c] (transposed stage)
    __shared__ bf16_t sO[128 * SO_STRIDE];     // output stage (union QK/V use)

    const int t    = threadIdx.x;
    const int lane = t & 31;
    const int wave = t >> 5;
    const int osub = wave & 3;
    const int ig   = wave >> 2;
    const int ot   = blockIdx.x >> 3;
    const int it   = blockIdx.x & 7;
    const int b    = blockIdx.y;
    const int o0   = ot * 64;                  // one full head per block
    const int i0   = it * 128;
    const int r16  = lane & 15;
    const int hi   = lane >> 4;
    const int kb   = hi * 8;
    const int h    = o0 >> 6;

    const float* Wmat[3] = { Wq, Wk, Wv };

    v8f acc[3][4] = {};

    for (int c0 = 0; c0 < CCH; c0 += 32) {
        // --- stage 3 W tiles: one ds_store_b128 per thread per matrix ---
        {
            int oo = t >> 2, cseg = (t & 3) * 8;
#pragma unroll
            for (int m = 0; m < 3; ++m) {
                const float* wr = Wmat[m] + (size_t)(o0 + oo) * CCH + c0 + cseg;
                v4f f0 = *(const v4f*)(wr);
                v4f f1 = *(const v4f*)(wr + 4);
                v8bf w8;
#pragma unroll
                for (int j = 0; j < 4; ++j) {
                    w8[j]     = (bf16_t)f0[j];
                    w8[j + 4] = (bf16_t)f1[j];
                }
                *(v8bf*)&sW[m][oo * SW_STRIDE + cseg] = w8;
            }
        }
        // --- stage x tile transposed, packing channel pairs (b32 stores) ---
#pragma unroll
        for (int rep = 0; rep < 8; ++rep) {
            int e  = rep * 256 + t;
            int cp = e >> 7;                   // 0..15 channel pair
            int ii = e & 127;
            const float* xr = x + ((size_t)b * CCH + c0 + 2 * cp) * NPIX + i0 + ii;
            v2bf p;
            p[0] = (bf16_t)xr[0];
            p[1] = (bf16_t)xr[NPIX];
            *(v2bf*)&sX[ii * SW_STRIDE + 2 * cp] = p;
        }
        if (c0 + 32 < CCH)
            __builtin_prefetch(x + ((size_t)b * CCH + c0 + 32) * NPIX + i0, 0, 1);
        __syncthreads();

        v16bf af[3];
#pragma unroll
        for (int m = 0; m < 3; ++m)
            af[m] = load_frag(&sW[m][(osub * 16 + r16) * SW_STRIDE + kb]);
#pragma unroll
        for (int s = 0; s < 4; ++s) {
            v16bf bf = load_frag(&sX[(ig * 64 + s * 16 + r16) * SW_STRIDE + kb]);
            acc[0][s] = wmma_bf16(af[0], bf, acc[0][s]);
            acc[1][s] = wmma_bf16(af[1], bf, acc[1][s]);
            acc[2][s] = wmma_bf16(af[2], bf, acc[2][s]);
        }
        __syncthreads();
    }

    // ---- epilogue: post-op -> sO -> coalesced global stores ----
    const size_t bh = (size_t)b * HEADS + h;

    // Q
#pragma unroll
    for (int s = 0; s < 4; ++s)
#pragma unroll
        for (int rr = 0; rr < 8; ++rr) {
            int ol = osub * 16 + rr + 8 * hi;
            int il = ig * 64 + s * 16 + r16;
            sO[il * SO_STRIDE + ol] = (bf16_t)((acc[0][s][rr] + bq[o0 + ol]) * 0.125f);
        }
    __syncthreads();
#pragma unroll
    for (int rep = 0; rep < 4; ++rep) {
        int ch = rep * 256 + t;
        int il = ch >> 3, seg = (ch & 7) * 8;
        *(v8bf*)&q_ws[(bh * NPIX + i0 + il) * DH + seg] =
            *(const v8bf*)&sO[il * SO_STRIDE + seg];
    }
    __syncthreads();

    // K  (+ folded relative-position bias: cc+cp == q^T (k + rel))
#pragma unroll
    for (int s = 0; s < 4; ++s)
#pragma unroll
        for (int rr = 0; rr < 8; ++rr) {
            int ol = osub * 16 + rr + 8 * hi;
            int il = ig * 64 + s * 16 + r16;
            int i  = i0 + il;
            float v = acc[1][s][rr] + bk[o0 + ol] +
                      rel_h[ol * 32 + (i >> 5)] + rel_w[ol * 32 + (i & 31)];
            sO[il * SO_STRIDE + ol] = (bf16_t)v;
        }
    __syncthreads();
#pragma unroll
    for (int rep = 0; rep < 4; ++rep) {
        int ch = rep * 256 + t;
        int il = ch >> 3, seg = (ch & 7) * 8;
        *(v8bf*)&k_ws[(bh * NPIX + i0 + il) * DH + seg] =
            *(const v8bf*)&sO[il * SO_STRIDE + seg];
    }
    __syncthreads();

    // V (transposed [dd][i] for contiguous PV B-fragments)
#pragma unroll
    for (int s = 0; s < 4; ++s)
#pragma unroll
        for (int rr = 0; rr < 8; ++rr) {
            int ol = osub * 16 + rr + 8 * hi;
            int il = ig * 64 + s * 16 + r16;
            sO[ol * SOV_STRIDE + il] = (bf16_t)(acc[2][s][rr] + bv[o0 + ol]);
        }
    __syncthreads();
#pragma unroll
    for (int rep = 0; rep < 4; ++rep) {
        int ch  = rep * 256 + t;
        int dd  = ch >> 4, seg = (ch & 15) * 8;
        *(v8bf*)&v_ws[(bh * DH + dd) * NPIX + i0 + seg] =
            *(const v8bf*)&sO[dd * SOV_STRIDE + seg];
    }
}

// ---------------------------------------------------------------------------
// Flash attention.  grid = (8 q-tiles, 128 b*h), block = 256 (8 waves x 16 q).
// K/V tiles double-buffered in LDS, DMA'd by the Tensor Data Mover: wave 0
// issues tile i+1 while all waves compute on tile i; TENSORcnt<=2 drains the
// current tile (in-order completion, 2 TDM ops per tile).
// ---------------------------------------------------------------------------
#define SK_STRIDE 72      // 64 + 8 halfs == TDM pad: 4 dwords every 32 dwords
#define SV_STRIDE 40      // 32 + 8 halfs == TDM pad: 4 dwords every 16 dwords
__global__ __launch_bounds__(256) void attn_kernel(
    const bf16_t* __restrict__ qw, const bf16_t* __restrict__ kw,
    const bf16_t* __restrict__ vw, float* __restrict__ out)
{
    __shared__ bf16_t sK[2][32 * SK_STRIDE];   // [buf][key][d]
    __shared__ bf16_t sV[2][64 * SV_STRIDE];   // [buf][dd][key] (V already ^T)
    __shared__ bf16_t sP[8][16 * SV_STRIDE];   // wave-private P tiles

    const int t    = threadIdx.x;
    const int lane = t & 31;
    const int wave = t >> 5;
    const int bh   = blockIdx.y;
    const int q0   = blockIdx.x * 128 + wave * 16;
    const int r16  = lane & 15;
    const int hi   = lane >> 4;
    const int kb   = hi * 8;

    const bf16_t* kwb = kw + (size_t)bh * NPIX * DH;
    const bf16_t* vwb = vw + (size_t)bh * DH * NPIX;

    const bf16_t* qbase = qw + (size_t)bh * NPIX * DH;
    v16bf qf0 = load_frag(qbase + (size_t)(q0 + r16) * DH + kb);
    v16bf qf1 = load_frag(qbase + (size_t)(q0 + r16) * DH + 32 + kb);

    v8f O[4] = {};
    float m[8], l[8];
#pragma unroll
    for (int rr = 0; rr < 8; ++rr) { m[rr] = -1e30f; l[rr] = 0.0f; }

    // prologue: DMA tile 0 into buffer 0
    if (wave == 0) {
        tdm_load_2d((u32)(size_t)&sK[0][0], kwb,
                    DH, 32, DH, 32, DH, /*pad 4dw/32dw*/4, 3);
        tdm_load_2d((u32)(size_t)&sV[0][0], vwb,
                    32, DH, 32, DH, NPIX, /*pad 4dw/16dw*/3, 3);
    }

    for (int k0 = 0; k0 < NPIX; k0 += 32) {
        const int cur = (k0 >> 5) & 1;
        if (wave == 0) {
            if (k0 + 32 < NPIX) {       // DMA next tile into the other buffer
                tdm_load_2d((u32)(size_t)&sK[cur ^ 1][0],
                            kwb + (size_t)(k0 + 32) * DH,
                            DH, 32, DH, 32, DH, 4, 3);
                tdm_load_2d((u32)(size_t)&sV[cur ^ 1][0],
                            vwb + (k0 + 32),
                            32, DH, 32, DH, NPIX, 3, 3);
                __builtin_amdgcn_s_wait_tensorcnt(2);   // tile k0 landed
            } else {
                __builtin_amdgcn_s_wait_tensorcnt(0);   // final drain
            }
        }
        __syncthreads();               // publish DMA'd tile to all waves

        // load all four B fragments first, then issue the WMMAs
        const bf16_t* kt = sK[cur];
        v16bf bkf[4];
        bkf[0] = load_frag(&kt[r16 * SK_STRIDE + kb]);
        bkf[1] = load_frag(&kt[r16 * SK_STRIDE + 32 + kb]);
        bkf[2] = load_frag(&kt[(16 + r16) * SK_STRIDE + kb]);
        bkf[3] = load_frag(&kt[(16 + r16) * SK_STRIDE + 32 + kb]);
        v8f Sa = {}, Sb = {};
        Sa = wmma_bf16(qf0, bkf[0], Sa);
        Sb = wmma_bf16(qf0, bkf[2], Sb);
        Sa = wmma_bf16(qf1, bkf[1], Sa);
        Sb = wmma_bf16(qf1, bkf[3], Sb);

        // online softmax (row stats replicated across each 16-lane half)
#pragma unroll
        for (int rr = 0; rr < 8; ++rr) {
            float a = Sa[rr], bv_ = Sb[rr];
            float tm = fmaxf(a, bv_);
            tm = fmaxf(tm, __shfl_xor(tm, 1));
            tm = fmaxf(tm, __shfl_xor(tm, 2));
            tm = fmaxf(tm, __shfl_xor(tm, 4));
            tm = fmaxf(tm, __shfl_xor(tm, 8));
            float nm   = fmaxf(m[rr], tm);
            float corr = __expf(m[rr] - nm);
            m[rr] = nm;
            float pa = __expf(a - nm), pb = __expf(bv_ - nm);
            Sa[rr] = pa; Sb[rr] = pb;
            float rs = pa + pb;
            rs += __shfl_xor(rs, 1);
            rs += __shfl_xor(rs, 2);
            rs += __shfl_xor(rs, 4);
            rs += __shfl_xor(rs, 8);
            l[rr] = l[rr] * corr + rs;
            O[0][rr] *= corr; O[1][rr] *= corr;
            O[2][rr] *= corr; O[3][rr] *= corr;
        }

        // C-layout P -> LDS -> A-layout (wave-private, DS pipeline in-order)
        bf16_t* pt = sP[wave];
#pragma unroll
        for (int rr = 0; rr < 8; ++rr) {
            int row = rr + 8 * hi;
            pt[row * SV_STRIDE + r16]      = (bf16_t)Sa[rr];
            pt[row * SV_STRIDE + 16 + r16] = (bf16_t)Sb[rr];
        }
        __builtin_amdgcn_wave_barrier();
        v16bf pf = load_frag(&pt[r16 * SV_STRIDE + kb]);

        const bf16_t* vt = sV[cur];
        v16bf bvf[4];
#pragma unroll
        for (int f = 0; f < 4; ++f)
            bvf[f] = load_frag(&vt[(f * 16 + r16) * SV_STRIDE + kb]);
#pragma unroll
        for (int f = 0; f < 4; ++f)
            O[f] = wmma_bf16(pf, bvf[f], O[f]);

        __syncthreads();  // buffer free before it is DMA'd again (i+2)
    }

    // normalize + store: out flat = [b][h][i][dd] == reference reshape
    float* obase = out + ((size_t)bh * NPIX + q0) * DH;
#pragma unroll
    for (int f = 0; f < 4; ++f) {
#pragma unroll
        for (int rr = 0; rr < 8; ++rr) {
            int qq = rr + 8 * hi;
            int dd = f * 16 + r16;
            obase[(size_t)qq * DH + dd] = O[f][rr] / l[rr];
        }
    }
}

// ---------------------------------------------------------------------------
extern "C" void kernel_launch(void* const* d_in, const int* in_sizes, int n_in,
                              void* d_out, int out_size, void* d_ws, size_t ws_size,
                              hipStream_t stream) {
    (void)in_sizes; (void)n_in; (void)out_size; (void)ws_size;
    const float* x     = (const float*)d_in[0];
    const float* Wq    = (const float*)d_in[1];
    const float* bq    = (const float*)d_in[2];
    const float* Wk    = (const float*)d_in[3];
    const float* bk    = (const float*)d_in[4];
    const float* Wv    = (const float*)d_in[5];
    const float* bv    = (const float*)d_in[6];
    const float* rel_h = (const float*)d_in[7];
    const float* rel_w = (const float*)d_in[8];

    const size_t elems = (size_t)BATCH * HEADS * NPIX * DH;  // 8M bf16 each
    bf16_t* q_ws = (bf16_t*)d_ws;
    bf16_t* k_ws = q_ws + elems;
    bf16_t* v_ws = k_ws + elems;

    qkv_proj_kernel<<<dim3(64, BATCH), 256, 0, stream>>>(
        x, Wq, bq, Wk, bk, Wv, bv, rel_h, rel_w, q_ws, k_ws, v_ws);

    attn_kernel<<<dim3(NPIX / 128, BATCH * HEADS), 256, 0, stream>>>(
        q_ws, k_ws, v_ws, (float*)d_out);
}
